// WordAttn_17506286698670
// MI455X (gfx1250) — matrix operand order repaired and checked by reading, not compiled, not run on hardware
//
#include <hip/hip_runtime.h>
#include <hip/hip_bf16.h>
#include <float.h>

// Problem constants (from reference): B=32, S=64, T=32, D=512, A=256
#define TT 32
#define DD 512
#define AA 256
#define HP 520   // padded LDS row stride in bf16 elems (mult of 8 -> 16B aligned reads)

typedef __attribute__((ext_vector_type(16))) __bf16 v16bf;
typedef __attribute__((ext_vector_type(2)))  __bf16 v2bf;
typedef __attribute__((ext_vector_type(8)))  float  v8f;

union FragBf { uint4 u[2]; v16bf v; };

// round-to-nearest-even f32 -> bf16 bit pattern (scalar fallback)
__device__ __forceinline__ unsigned short f2bf(float x) {
    unsigned u = __float_as_uint(x);
    unsigned r = u + 0x7FFFu + ((u >> 16) & 1u);
    return (unsigned short)(r >> 16);
}

// packed f32x2 -> bf16x2 (v_cvt_pk_bf16_f32 on gfx1250 when available)
__device__ __forceinline__ unsigned pk2bf(float x, float y) {
#if __has_builtin(__builtin_amdgcn_cvt_pk_bf16_f32)
    union { v2bf v; unsigned u; } c;
    c.v = __builtin_amdgcn_cvt_pk_bf16_f32(x, y);
    return c.u;
#else
    return (unsigned)f2bf(x) | ((unsigned)f2bf(y) << 16);
#endif
}

// native v_tanh_f32 trans-op when available
__device__ __forceinline__ float fast_tanh(float x) {
#if __has_builtin(__builtin_amdgcn_tanhf)
    return __builtin_amdgcn_tanhf(x);
#else
    return tanhf(x);
#endif
}

// --------------------------------------------------------------------------
// Prep: swizzle w_weight [A=256][D=512] f32 -> bf16 B-fragment layout.
// Fragment f = nt*16 + ks (nt: N-tile 0..15, ks: K-step 0..15).
// Within a fragment: lane l (0..31), elem j (0..15):
//   value = w[nt*16 + (l&15)][ks*32 + (l>>4)*8 + (j&7) + (j>>3)*16]
// Stored contiguously: idx = (f*32 + l)*16 + j   (bf16 elements)
// --------------------------------------------------------------------------
__global__ __launch_bounds__(256) void wattn_prep_bswz(
    const float* __restrict__ w, unsigned short* __restrict__ Bswz)
{
    int idx = blockIdx.x * 256 + threadIdx.x;   // 0 .. 131071
    int f   = idx >> 9;
    int rem = idx & 511;
    int l   = rem >> 4;
    int j   = rem & 15;
    int n   = ((f >> 4) << 4) + (l & 15);
    int k   = ((f & 15) << 5) + ((l >> 4) << 3) + (j & 7) + ((j >> 3) << 4);
    Bswz[idx] = f2bf(w[n * DD + k]);
}

// --------------------------------------------------------------------------
// Main fused kernel: one block per (b,s); 256 threads = 8 waves.
// --------------------------------------------------------------------------
__global__ __launch_bounds__(256) void wattn_main(
    const float* __restrict__ H,            // [B,S,T,D] f32
    const unsigned char* __restrict__ mask, // [B,S,T] bool (1 byte)
    const float* __restrict__ bias,         // [A]
    const float* __restrict__ u,            // [A]
    const unsigned short* __restrict__ Bswz,// pre-swizzled bf16 w
    float* __restrict__ out)                // [B,S,D]
{
    __shared__ unsigned short Hs[TT * HP];  // 33,280 B bf16 H tile
    __shared__ float e_part[8][TT];         // per-wave logit partials
    __shared__ float w_sh[TT];              // softmax weights

    const int tid  = threadIdx.x;
    const int lane = tid & 31;
    const int wv   = tid >> 5;
    const int bs   = blockIdx.x;
    const float* Hb = H + (size_t)bs * (TT * DD);

    // ---- stage H tile: f32 global -> bf16 LDS (coalesced float4 loads) ----
    const float4* H4 = (const float4*)Hb;   // 4096 float4s
    #pragma unroll
    for (int i = 0; i < 16; ++i) {
        int j = i * 256 + tid;
        float4 v = H4[j];
        int e0 = j << 2;                    // flat element index
        int t  = e0 >> 9;
        int c  = e0 & 511;
        uint2 pk;
        pk.x = pk2bf(v.x, v.y);
        pk.y = pk2bf(v.z, v.w);
        *(uint2*)&Hs[t * HP + c] = pk;
    }
    __syncthreads();

    // ---- WMMA GEMM: wave handles N-tiles {2wv, 2wv+1}, both M-tiles ----
    const int half = lane >> 4;
    const int l16  = lane & 15;
    const int nt0  = wv * 2;
    const int nt1  = nt0 + 1;

    v8f acc00 = {}, acc01 = {}, acc10 = {}, acc11 = {};  // [m][n]

    #pragma unroll
    for (int ks = 0; ks < 16; ++ks) {
        const int k0 = ks << 5;
        // A fragments (16-bit A 16x32 layout): lane group half -> K ofs half*8,
        // elems 0..7 at k0+half*8, elems 8..15 at k0+16+half*8
        FragBf a0, a1;
        const unsigned short* r0 = &Hs[l16 * HP + k0 + half * 8];
        a0.u[0] = *(const uint4*)(r0);
        a0.u[1] = *(const uint4*)(r0 + 16);
        const unsigned short* r1 = r0 + 16 * HP;
        a1.u[0] = *(const uint4*)(r1);
        a1.u[1] = *(const uint4*)(r1 + 16);
        // B fragments: pre-swizzled, contiguous 32B per lane
        FragBf b0, b1;
        const uint4* pb0 = (const uint4*)(Bswz + (((size_t)(nt0 * 16 + ks) * 32 + lane) << 4));
        b0.u[0] = pb0[0]; b0.u[1] = pb0[1];
        const uint4* pb1 = (const uint4*)(Bswz + (((size_t)(nt1 * 16 + ks) * 32 + lane) << 4));
        b1.u[0] = pb1[0]; b1.u[1] = pb1[1];

        acc00 = __builtin_amdgcn_wmma_f32_16x16x32_bf16(false, a0.v, false, b0.v, (short)0, acc00, false, false);
        acc01 = __builtin_amdgcn_wmma_f32_16x16x32_bf16(false, a0.v, false, b1.v, (short)0, acc01, false, false);
        acc10 = __builtin_amdgcn_wmma_f32_16x16x32_bf16(false, a1.v, false, b0.v, (short)0, acc10, false, false);
        acc11 = __builtin_amdgcn_wmma_f32_16x16x32_bf16(false, a1.v, false, b1.v, (short)0, acc11, false, false);
    }

    // ---- epilogue: e[t] += sum_n tanh(acc + bias[n]) * u[n] ----
    const int n0 = (nt0 << 4) + l16;
    const int n1 = n0 + 16;
    const float bs0 = bias[n0], bs1 = bias[n1];
    const float uu0 = u[n0],    uu1 = u[n1];

    #pragma unroll
    for (int mt = 0; mt < 2; ++mt) {
        v8f A0 = mt ? acc10 : acc00;
        v8f A1 = mt ? acc11 : acc01;
        #pragma unroll
        for (int r = 0; r < 8; ++r) {
            // C/D layout: VGPR r, lanes 0-15 -> M = mt*16+r, lanes 16-31 -> M = mt*16+8+r
            float p = fast_tanh(A0[r] + bs0) * uu0 + fast_tanh(A1[r] + bs1) * uu1;
            #pragma unroll
            for (int off = 1; off < 16; off <<= 1)
                p += __shfl_xor(p, off, 32);
            if (l16 == 0)
                e_part[wv][(mt << 4) + (half << 3) + r] = p;
        }
    }
    __syncthreads();

    // ---- deterministic reduce + masked softmax over T=32 (wave 0) ----
    if (wv == 0) {
        float e = 0.f;
        #pragma unroll
        for (int k = 0; k < 8; ++k) e += e_part[k][lane];
        bool mk = mask[(size_t)bs * TT + lane] != 0;
        float x = mk ? e : -FLT_MAX;
        float mx = x;
        #pragma unroll
        for (int off = 1; off < 32; off <<= 1)
            mx = fmaxf(mx, __shfl_xor(mx, off, 32));
        float ex = __expf(x - mx);
        float sm = ex;
        #pragma unroll
        for (int off = 1; off < 32; off <<= 1)
            sm += __shfl_xor(sm, off, 32);
        w_sh[lane] = ex / sm;
    }
    __syncthreads();

    // ---- weighted sum over T in full f32 (H tile is cache-hot) ----
    float s0 = 0.f, s1 = 0.f;
    #pragma unroll
    for (int t = 0; t < TT; ++t) {
        float wt = w_sh[t];
        s0 = fmaf(wt, Hb[t * DD + tid], s0);
        s1 = fmaf(wt, Hb[t * DD + tid + 256], s1);
    }
    out[(size_t)bs * DD + tid]       = s0;
    out[(size_t)bs * DD + tid + 256] = s1;
}

// --------------------------------------------------------------------------
extern "C" void kernel_launch(void* const* d_in, const int* in_sizes, int n_in,
                              void* d_out, int out_size, void* d_ws, size_t ws_size,
                              hipStream_t stream) {
    const float*         H      = (const float*)d_in[0];
    const unsigned char* msk    = (const unsigned char*)d_in[1];
    const float*         w_w    = (const float*)d_in[2];
    const float*         w_b    = (const float*)d_in[3];
    const float*         u_w    = (const float*)d_in[4];
    float*               out    = (float*)d_out;
    unsigned short*      Bswz   = (unsigned short*)d_ws;   // 256 KB bf16 swizzled w

    // 1) swizzle + convert weights (A*D = 131072 elements)
    wattn_prep_bswz<<<(AA * DD) / 256, 256, 0, stream>>>(w_w, Bswz);
    // 2) fused attention: one block per (b,s) = 2048 blocks
    wattn_main<<<32 * 64, 256, 0, stream>>>(H, msk, w_b, u_w, Bswz, out);
}
